// TransformerLayer_53566832115756
// MI455X (gfx1250) — matrix operand order repaired and checked by reading, not compiled, not run on hardware
//
#include <hip/hip_runtime.h>
#include <hip/hip_bf16.h>

#define N_TOK 65536
#define DIM   512
#define NH    8
#define HDIM  64
#define FFD   2048
#define EPSV  1e-5f
#define MTILE 32            // tokens per workgroup (2 WMMA row-tiles)

typedef __attribute__((ext_vector_type(16))) __bf16 v16bf;
typedef __attribute__((ext_vector_type(8)))  float  v8f;

union V16U { v16bf v; int4 q[2]; };

__device__ __forceinline__ unsigned short f2bf(float f) {
    unsigned int u = __float_as_uint(f);
    unsigned int r = (u + 0x7FFFu + ((u >> 16) & 1u)) >> 16;  // RNE
    return (unsigned short)r;
}

// A fragment (16x32 bf16, MxK) from LDS, row-major [row][k], per CDNA5 layout:
// elems 0..7  -> K = kbase + half*8 + e
// elems 8..15 -> K = kbase + 16 + half*8 + (e-8)
__device__ __forceinline__ v16bf load_A_lds(const unsigned short* base, int row,
                                            int stride, int kbase, int half) {
    const unsigned short* p = base + row * stride + kbase + half * 8;
    V16U u;
    u.q[0] = *(const int4*)(p);
    u.q[1] = *(const int4*)(p + 16);
    return u.v;
}

// B fragment (32x16 bf16, KxN) from global, stored transposed [n][k]:
// lane n = lane&15, elems 0..7 -> K = kbase + half*8 + e, elems 8..15 -> K = kbase + 16 + half*8 + (e-8)
__device__ __forceinline__ v16bf load_B_glb(const unsigned short* WT, int ncol,
                                            int Kdim, int kbase, int half) {
    const unsigned short* p = WT + (size_t)ncol * Kdim + kbase + half * 8;
    V16U u;
    u.q[0] = *(const int4*)(p);
    u.q[1] = *(const int4*)(p + 16);
    return u.v;
}

#define WMMA_BF16(a, b, c) \
    __builtin_amdgcn_wmma_f32_16x16x32_bf16(false, (a), false, (b), (short)0, (c), false, false)

// ---------------------------------------------------------------------------
// K1: pack weights to bf16, transposed for contiguous B-fragment loads.
//   WqkvT [1536][512], W1T [2048][512], W2T [512][2048], bqkv[1536] f32
// ---------------------------------------------------------------------------
__global__ void pack_weights(const float* __restrict__ Wq, const float* __restrict__ Wk,
                             const float* __restrict__ Wv, const float* __restrict__ W1,
                             const float* __restrict__ W2,
                             const float* __restrict__ bq, const float* __restrict__ bk,
                             const float* __restrict__ bv,
                             unsigned short* __restrict__ WqkvT,
                             unsigned short* __restrict__ W1T,
                             unsigned short* __restrict__ W2T,
                             float* __restrict__ bqkv) {
    const int stride = gridDim.x * blockDim.x;
    const int g = blockIdx.x * blockDim.x + threadIdx.x;
    for (int idx = g; idx < 3 * DIM * DIM; idx += stride) {
        int n = idx / DIM, k = idx % DIM;
        const float* W = (n < DIM) ? Wq : ((n < 2 * DIM) ? Wk : Wv);
        WqkvT[idx] = f2bf(W[k * DIM + (n & (DIM - 1))]);
    }
    for (int idx = g; idx < FFD * DIM; idx += stride) {
        int f = idx / DIM, d = idx % DIM;
        W1T[idx] = f2bf(W1[d * FFD + f]);
    }
    for (int idx = g; idx < DIM * FFD; idx += stride) {
        int d = idx / FFD, f = idx % FFD;
        W2T[idx] = f2bf(W2[f * DIM + d]);
    }
    for (int n = g; n < 3 * DIM; n += stride) {
        const float* bb = (n < DIM) ? bq : ((n < 2 * DIM) ? bk : bv);
        bqkv[n] = bb[n & (DIM - 1)];
    }
}

// ---------------------------------------------------------------------------
// K2: fused QKV GEMM + per-token head attention + residual + LN1.
// 512 threads (16 waves), 32 tokens per workgroup (2 row-tiles -> each B
// fragment feeds 2 WMMAs).
// LDS: x f32 (64K) | x bf16 (32K) | qkv f32 (192K) | scores (8K) | red (~4K)
// ---------------------------------------------------------------------------
#define K2_SX    0
#define K2_SXB   (K2_SX + MTILE * DIM * 4)            // 65536
#define K2_SQKV  (K2_SXB + MTILE * DIM * 2)           // + 32768
#define K2_SS    (K2_SQKV + MTILE * 1536 * 4)         // + 196608
#define K2_SRED  (K2_SS + MTILE * NH * NH * 4)        // + 8192
#define K2_SMV   (K2_SRED + MTILE * 16 * 2 * 4)       // + 4096
#define K2_SMEM  (K2_SMV + MTILE * 2 * 4)             // + 256  => 307456 B

__global__ __launch_bounds__(512) void attn_block(
    const float* __restrict__ x, const unsigned short* __restrict__ WqkvT,
    const float* __restrict__ bqkv,
    const float* __restrict__ ln1g, const float* __restrict__ ln1b,
    float* __restrict__ yf, unsigned short* __restrict__ yb) {
    extern __shared__ char smem[];
    float*          sx   = (float*)(smem + K2_SX);
    unsigned short* sxb  = (unsigned short*)(smem + K2_SXB);
    float*          sqkv = (float*)(smem + K2_SQKV);
    float*          ss   = (float*)(smem + K2_SS);
    float*          sred = (float*)(smem + K2_SRED);
    float*          smv  = (float*)(smem + K2_SMV);

    const int t = threadIdx.x;
    const int tok0 = blockIdx.x * MTILE;

    // stage x tile (f32 for residual, bf16 for WMMA A)
    for (int i = 0; i < 32; ++i) {
        int idx = t + i * 512;
        float v = x[(size_t)tok0 * DIM + idx];
        sx[idx] = v;
        sxb[idx] = f2bf(v);
    }
    __syncthreads();

    const int lane = t & 31, wv = t >> 5;
    const int half = lane >> 4, mrow = lane & 15;

    // ---- QKV GEMM: 32 x 1536, K = 512; 6 col-tiles x 2 row-tiles per wave ----
    v8f acc0[6] = {}, acc1[6] = {};
    for (int kb = 0; kb < DIM; kb += 32) {
        v16bf a0 = load_A_lds(sxb, mrow, DIM, kb, half);
        v16bf a1 = load_A_lds(sxb, 16 + mrow, DIM, kb, half);
#pragma unroll
        for (int tt = 0; tt < 6; ++tt) {
            int ncol = (wv * 6 + tt) * 16 + mrow;
            v16bf b = load_B_glb(WqkvT, ncol, DIM, kb, half);
            acc0[tt] = WMMA_BF16(a0, b, acc0[tt]);
            acc1[tt] = WMMA_BF16(a1, b, acc1[tt]);
        }
    }
    // bias + spill qkv tile to LDS (C layout: M = r + 8*half, N = lane&15)
    for (int tt = 0; tt < 6; ++tt) {
        int n = (wv * 6 + tt) * 16 + mrow;
        float bias = bqkv[n];
#pragma unroll
        for (int r = 0; r < 8; ++r) {
            int m = r + half * 8;
            sqkv[m * 1536 + n] = acc0[tt][r] + bias;
            sqkv[(16 + m) * 1536 + n] = acc1[tt][r] + bias;
        }
    }
    __syncthreads();

    // ---- scores: s[m][h][g] = q.k / sqrt(64) ----
    {
        int m = t >> 4, idx = t & 15;          // 32 tokens x 16 workers
        int h = idx >> 1, g0 = (idx & 1) * 4;
        const float* qv = sqkv + m * 1536 + h * HDIM;
        for (int g = g0; g < g0 + 4; ++g) {
            const float* kv = sqkv + m * 1536 + DIM + g * HDIM;
            float s = 0.f;
            for (int d = 0; d < HDIM; ++d) s += qv[d] * kv[d];
            ss[(m * NH + h) * NH + g] = s * 0.125f;
        }
    }
    __syncthreads();
    // ---- softmax over g (8 values) ----
    if (t < MTILE * NH) {
        int m = t >> 3, h = t & 7;
        float* sp = ss + (m * NH + h) * NH;
        float mx = sp[0];
        for (int g = 1; g < NH; ++g) mx = fmaxf(mx, sp[g]);
        float e[NH], sum = 0.f;
        for (int g = 0; g < NH; ++g) { e[g] = __expf(sp[g] - mx); sum += e[g]; }
        float inv = 1.f / sum;
        for (int g = 0; g < NH; ++g) sp[g] = e[g] * inv;
    }
    __syncthreads();
    // ---- ctx + residual into sx ----
    for (int i = 0; i < 32; ++i) {
        int idx = t + i * 512;
        int m = idx >> 9, c = idx & (DIM - 1), h = c >> 6, d = c & (HDIM - 1);
        const float* ap = ss + (m * NH + h) * NH;
        const float* vp = sqkv + m * 1536 + 2 * DIM + d;
        float ctx = 0.f;
#pragma unroll
        for (int g = 0; g < NH; ++g) ctx += ap[g] * vp[g * HDIM];
        sx[idx] += ctx;
    }
    __syncthreads();

    // ---- LN1: 16 threads per token ----
    {
        int m = t >> 4, sub = t & 15;
        float s = 0.f, s2 = 0.f;
        for (int i = 0; i < 32; ++i) {
            float v = sx[m * DIM + sub + i * 16];
            s += v; s2 += v * v;
        }
        sred[(m * 16 + sub) * 2] = s;
        sred[(m * 16 + sub) * 2 + 1] = s2;
    }
    __syncthreads();
    if ((t & 15) == 0) {
        int m = t >> 4;
        float s = 0.f, s2 = 0.f;
        for (int i = 0; i < 16; ++i) { s += sred[(m * 16 + i) * 2]; s2 += sred[(m * 16 + i) * 2 + 1]; }
        float mu = s * (1.f / DIM);
        float var = s2 * (1.f / DIM) - mu * mu;
        smv[m * 2] = mu;
        smv[m * 2 + 1] = rsqrtf(var + EPSV);
    }
    __syncthreads();
    {
        int m = t >> 4, sub = t & 15;
        float mu = smv[m * 2], rs = smv[m * 2 + 1];
        for (int i = 0; i < 32; ++i) {
            int c = sub + i * 16;
            float v = (sx[m * DIM + c] - mu) * rs * ln1g[c] + ln1b[c];
            size_t o = (size_t)(tok0 + m) * DIM + c;
            yf[o] = v;
            yb[o] = f2bf(v);
        }
    }
}

// ---------------------------------------------------------------------------
// K3: fused FFN: h = relu(y@W1+b1) kept in LDS; z = h@W2 + b2 + y; LN2.
// 512 threads (16 waves), 32 tokens per workgroup (2 row-tiles).
// LDS: y bf16 (32K) | y f32 (64K) | h bf16 (128K, reused as z f32) | red
// ---------------------------------------------------------------------------
#define K3_SYB   0
#define K3_SY    (K3_SYB + MTILE * DIM * 2)           // 32768
#define K3_SH    (K3_SY + MTILE * DIM * 4)            // + 65536
#define K3_SRED  (K3_SH + MTILE * FFD * 2)            // + 131072
#define K3_SMV   (K3_SRED + MTILE * 16 * 2 * 4)       // + 4096
#define K3_SMEM  (K3_SMV + MTILE * 2 * 4)             // + 256  => 233728 B

__global__ __launch_bounds__(512) void ffn_block(
    const float* __restrict__ yf, const unsigned short* __restrict__ yb,
    const unsigned short* __restrict__ W1T, const float* __restrict__ b1,
    const unsigned short* __restrict__ W2T, const float* __restrict__ b2,
    const float* __restrict__ ln2g, const float* __restrict__ ln2b,
    float* __restrict__ out) {
    extern __shared__ char smem[];
    unsigned short* syb  = (unsigned short*)(smem + K3_SYB);
    float*          sy   = (float*)(smem + K3_SY);
    unsigned short* sh   = (unsigned short*)(smem + K3_SH);
    float*          sz   = (float*)(smem + K3_SH);     // reuse h region as z f32
    float*          sred = (float*)(smem + K3_SRED);
    float*          smv  = (float*)(smem + K3_SMV);

    const int t = threadIdx.x;
    const int tok0 = blockIdx.x * MTILE;

    for (int i = 0; i < 32; ++i) {
        int idx = t + i * 512;
        size_t o = (size_t)tok0 * DIM + idx;
        syb[idx] = yb[o];
        sy[idx]  = yf[o];
    }
    __syncthreads();

    const int lane = t & 31, wv = t >> 5;
    const int half = lane >> 4, mrow = lane & 15;

    // ---- GEMM1: 32 x 2048, K = 512; 8 col-tiles x 2 row-tiles per wave ----
    v8f acc0[8] = {}, acc1[8] = {};
    for (int kb = 0; kb < DIM; kb += 32) {
        v16bf a0 = load_A_lds(syb, mrow, DIM, kb, half);
        v16bf a1 = load_A_lds(syb, 16 + mrow, DIM, kb, half);
#pragma unroll
        for (int tt = 0; tt < 8; ++tt) {
            int f = (wv * 8 + tt) * 16 + mrow;
            v16bf b = load_B_glb(W1T, f, DIM, kb, half);
            acc0[tt] = WMMA_BF16(a0, b, acc0[tt]);
            acc1[tt] = WMMA_BF16(a1, b, acc1[tt]);
        }
    }
    // bias + relu -> h tile in LDS (bf16)
    for (int tt = 0; tt < 8; ++tt) {
        int f = (wv * 8 + tt) * 16 + mrow;
        float bias = b1[f];
#pragma unroll
        for (int r = 0; r < 8; ++r) {
            int m = r + half * 8;
            sh[m * FFD + f] = f2bf(fmaxf(acc0[tt][r] + bias, 0.f));
            sh[(16 + m) * FFD + f] = f2bf(fmaxf(acc1[tt][r] + bias, 0.f));
        }
    }
    __syncthreads();

    // ---- GEMM2: 32 x 512, K = 2048; 2 col-tiles x 2 row-tiles per wave ----
    v8f c0[2] = {}, c1[2] = {};
    for (int kb = 0; kb < FFD; kb += 32) {
        v16bf a0 = load_A_lds(sh, mrow, FFD, kb, half);
        v16bf a1 = load_A_lds(sh, 16 + mrow, FFD, kb, half);
#pragma unroll
        for (int tt = 0; tt < 2; ++tt) {
            int n = (wv * 2 + tt) * 16 + mrow;
            v16bf b = load_B_glb(W2T, n, FFD, kb, half);
            c0[tt] = WMMA_BF16(a0, b, c0[tt]);
            c1[tt] = WMMA_BF16(a1, b, c1[tt]);
        }
    }
    __syncthreads();   // all waves done reading sh before we overwrite as z
    for (int tt = 0; tt < 2; ++tt) {
        int n = (wv * 2 + tt) * 16 + mrow;
        float bias = b2[n];
#pragma unroll
        for (int r = 0; r < 8; ++r) {
            int m = r + half * 8;
            sz[m * DIM + n] = c0[tt][r] + bias + sy[m * DIM + n];
            sz[(16 + m) * DIM + n] = c1[tt][r] + bias + sy[(16 + m) * DIM + n];
        }
    }
    __syncthreads();

    // ---- LN2: 16 threads per token ----
    {
        int m = t >> 4, sub = t & 15;
        float s = 0.f, s2 = 0.f;
        for (int i = 0; i < 32; ++i) {
            float v = sz[m * DIM + sub + i * 16];
            s += v; s2 += v * v;
        }
        sred[(m * 16 + sub) * 2] = s;
        sred[(m * 16 + sub) * 2 + 1] = s2;
    }
    __syncthreads();
    if ((t & 15) == 0) {
        int m = t >> 4;
        float s = 0.f, s2 = 0.f;
        for (int i = 0; i < 16; ++i) { s += sred[(m * 16 + i) * 2]; s2 += sred[(m * 16 + i) * 2 + 1]; }
        float mu = s * (1.f / DIM);
        float var = s2 * (1.f / DIM) - mu * mu;
        smv[m * 2] = mu;
        smv[m * 2 + 1] = rsqrtf(var + EPSV);
    }
    __syncthreads();
    {
        int m = t >> 4, sub = t & 15;
        float mu = smv[m * 2], rs = smv[m * 2 + 1];
        for (int i = 0; i < 32; ++i) {
            int c = sub + i * 16;
            float v = (sz[m * DIM + c] - mu) * rs * ln2g[c] + ln2b[c];
            out[(size_t)(tok0 + m) * DIM + c] = v;
        }
    }
}

// ---------------------------------------------------------------------------
extern "C" void kernel_launch(void* const* d_in, const int* in_sizes, int n_in,
                              void* d_out, int out_size, void* d_ws, size_t ws_size,
                              hipStream_t stream) {
    (void)in_sizes; (void)n_in; (void)out_size; (void)ws_size;
    const float* x   = (const float*)d_in[0];
    const float* Wq  = (const float*)d_in[1];
    const float* bq  = (const float*)d_in[2];
    const float* Wk  = (const float*)d_in[3];
    const float* bk  = (const float*)d_in[4];
    const float* Wv  = (const float*)d_in[5];
    const float* bv  = (const float*)d_in[6];
    const float* g1  = (const float*)d_in[7];
    const float* b1l = (const float*)d_in[8];
    const float* W1  = (const float*)d_in[9];
    const float* b1  = (const float*)d_in[10];
    const float* W2  = (const float*)d_in[11];
    const float* b2  = (const float*)d_in[12];
    const float* g2  = (const float*)d_in[13];
    const float* b2l = (const float*)d_in[14];
    float* out = (float*)d_out;

    char* ws = (char*)d_ws;
    size_t off = 0;
    auto carve = [&](size_t bytes) {
        char* p = ws + off;
        off = (off + bytes + 255) & ~(size_t)255;
        return p;
    };
    unsigned short* WqkvT = (unsigned short*)carve((size_t)3 * DIM * DIM * 2);
    unsigned short* W1T   = (unsigned short*)carve((size_t)FFD * DIM * 2);
    unsigned short* W2T   = (unsigned short*)carve((size_t)DIM * FFD * 2);
    float*          bqkv  = (float*)carve((size_t)3 * DIM * 4);
    float*          yf    = (float*)carve((size_t)N_TOK * DIM * 4);
    unsigned short* ybuf  = (unsigned short*)carve((size_t)N_TOK * DIM * 2);

    pack_weights<<<512, 256, 0, stream>>>(Wq, Wk, Wv, W1, W2, bq, bk, bv,
                                          WqkvT, W1T, W2T, bqkv);
    attn_block<<<N_TOK / MTILE, 512, K2_SMEM, stream>>>(x, WqkvT, bqkv, g1, b1l, yf, ybuf);
    ffn_block<<<N_TOK / MTILE, 512, K3_SMEM, stream>>>(yf, ybuf, W1T, b1, W2T, b2,
                                                       g2, b2l, out);
}
